// TransformerEncoderLayer_1915555414213
// MI455X (gfx1250) — compile-verified, matrix operand order checked by reading, not run on hardware
//
#include <hip/hip_runtime.h>

// ---------------------------------------------------------------------------
// TransformerEncoderLayer for MI455X (gfx1250, wave32, WMMA)
// B=32 D=16 M=128 E=512 H=8 F=2048 -> N = B*D*M = 65536 token rows.
// All GEMMs run through v_wmma_f32_16x16x32_f16 (f16 in, f32 accumulate).
// GEMM: block tile 128x256, wave tile 64x64 -> 16 WMMAs per 8 fragment loads
// (~32 FLOP per LDS byte), sized so 4 SIMDs of a WGP don't starve on LDS BW.
// ---------------------------------------------------------------------------

typedef __attribute__((ext_vector_type(16))) _Float16 v16h;
typedef __attribute__((ext_vector_type(8)))  _Float16 v8h;
typedef __attribute__((ext_vector_type(4)))  _Float16 v4h;
typedef __attribute__((ext_vector_type(8)))  float    v8f;

#define NROWS   65536   // B*D*M
#define EDIM    512
#define FDIM    2048
#define NSEQ    512     // B*D
#define MSG     128     // M
#define NHEAD   8
#define DK      64

// ---------------------------------------------------------------------------
// Fragment load from LDS, row-major storage with 16-bit elements.
// CDNA5 16-bit A-matrix 16x32 layout: lane L holds row (L%16); element e maps
// to K = (e/8)*16 + (L/16)*8 + (e%8).  B (32x16) stored as its transpose
// (column per lane) uses the identical pattern.  -> two contiguous 8-half
// (16-byte) LDS reads per fragment.
// ---------------------------------------------------------------------------
__device__ inline v16h ld_frag(const _Float16* rowp, int kb, int hi) {
    const _Float16* p = rowp + kb + (hi << 3);
    v8h lo = *(const v8h*)(p);
    v8h hh = *(const v8h*)(p + 16);
    v16h f;
#pragma unroll
    for (int i = 0; i < 8; ++i) { f[i] = lo[i]; f[i + 8] = hh[i]; }
    return f;
}

// 4-element global->LDS loaders with f32->f16 conversion when needed
__device__ inline void load4_to_lds(const float* g, _Float16* s) {
    float4 f = *(const float4*)g;
    s[0] = (_Float16)f.x; s[1] = (_Float16)f.y;
    s[2] = (_Float16)f.z; s[3] = (_Float16)f.w;
}
__device__ inline void load4_to_lds(const _Float16* g, _Float16* s) {
    v4h f = *(const v4h*)g;
    s[0] = f[0]; s[1] = f[1]; s[2] = f[2]; s[3] = f[3];
}

// ---------------------------------------------------------------------------
// Generic WMMA GEMM:  C[Nrows x Ncols] = act(A[Nrows x K] @ W[K x Ncols] + b)
// Block tile 128x256, BK=32, 256 threads (8 waves, 2x4), wave tile 64x64.
// Requires Ncols % 256 == 0, K % 32 == 0, Nrows % 128 == 0.
// ---------------------------------------------------------------------------
template <typename AT, bool RELU, typename OT>
__global__ void __launch_bounds__(256)
gemm_wmma(const AT* __restrict__ A, const float* __restrict__ W,
          const float* __restrict__ bias, OT* __restrict__ C,
          int K, int Ncols) {
    __shared__ alignas(16) _Float16 As[128 * 40];   // [row][k]  stride 40 halfs
    __shared__ alignas(16) _Float16 Bs[256 * 40];   // [col][k]  (W transposed)

    const int tid  = threadIdx.x;
    const int lane = tid & 31;
    const int wave = tid >> 5;
    const int lr   = lane & 15;
    const int hi   = lane >> 4;
    const int wm   = wave >> 2;       // 0..1 -> 64-row band
    const int wn   = wave & 3;        // 0..3 -> 64-col band
    const long rowBase = (long)blockIdx.x * 128;
    const int  colBase = blockIdx.y * 256;

    v8f acc[4][4];
#pragma unroll
    for (int i = 0; i < 4; ++i)
#pragma unroll
        for (int j = 0; j < 4; ++j)
#pragma unroll
            for (int t = 0; t < 8; ++t) acc[i][j][t] = 0.0f;

    for (int kb = 0; kb < K; kb += 32) {
        // Stage A tile (128x32), coalesced rows, convert to f16
#pragma unroll
        for (int c = 0; c < 4; ++c) {
            int idx = tid + c * 256;              // 0..1023
            int r   = idx >> 3;
            int sg  = (idx & 7) * 4;
            load4_to_lds(A + (rowBase + r) * (long)K + kb + sg, &As[r * 40 + sg]);
        }
        // Stage W tile (32x256) transposed into Bs[col][k]
#pragma unroll
        for (int c = 0; c < 8; ++c) {
            int idx = tid + c * 256;              // 0..2047
            int kr  = idx >> 6;                   // 0..31
            int sg  = (idx & 63) * 4;             // 0..252
            float4 f = *(const float4*)(W + (long)(kb + kr) * Ncols + colBase + sg);
            Bs[(sg + 0) * 40 + kr] = (_Float16)f.x;
            Bs[(sg + 1) * 40 + kr] = (_Float16)f.y;
            Bs[(sg + 2) * 40 + kr] = (_Float16)f.z;
            Bs[(sg + 3) * 40 + kr] = (_Float16)f.w;
        }
        // Prefetch next K tile toward the WGP (global_prefetch_b8)
        if (kb + 32 < K) {
            __builtin_prefetch(A + (rowBase + (tid >> 1)) * (long)K + kb + 32, 0, 0);
            __builtin_prefetch(W + (long)(kb + 32 + (tid & 31)) * Ncols + colBase
                                 + ((tid >> 5) << 5), 0, 0);
        }
        __syncthreads();

        v16h af[4];
#pragma unroll
        for (int i = 0; i < 4; ++i)
            af[i] = ld_frag(&As[(wm * 64 + i * 16 + lr) * 40], 0, hi);
#pragma unroll
        for (int j = 0; j < 4; ++j) {
            v16h bf = ld_frag(&Bs[(wn * 64 + j * 16 + lr) * 40], 0, hi);
#pragma unroll
            for (int i = 0; i < 4; ++i)
                acc[i][j] = __builtin_amdgcn_wmma_f32_16x16x32_f16(
                    false, af[i], false, bf, (short)0, acc[i][j], false, false);
        }
        __syncthreads();
    }

    // Epilogue: C layout -> lane lr = N, element v with hi -> M = hi*8 + v
#pragma unroll
    for (int j = 0; j < 4; ++j) {
        int   gc = colBase + wn * 64 + j * 16 + lr;
        float bv = bias[gc];
#pragma unroll
        for (int i = 0; i < 4; ++i) {
            long gr0 = rowBase + wm * 64 + i * 16 + hi * 8;
#pragma unroll
            for (int v = 0; v < 8; ++v) {
                float val = acc[i][j][v] + bv;
                if (RELU) val = fmaxf(val, 0.0f);
                C[(gr0 + v) * (long)Ncols + gc] = (OT)val;
            }
        }
    }
}

// ---------------------------------------------------------------------------
// Fused attention: one workgroup per (seq, head).  q,k,v^T staged f16 in LDS;
// scores (128x128) held in registers; masked softmax via half-wave shuffles;
// probabilities re-staged in LDS (overlaying dead q/k); P @ V via WMMA.
// ---------------------------------------------------------------------------
__global__ void __launch_bounds__(256)
attn_wmma(const _Float16* __restrict__ Q, const _Float16* __restrict__ Km,
          const _Float16* __restrict__ V, const int* __restrict__ mask,
          _Float16* __restrict__ O) {
    const int slice = blockIdx.x;          // 0..NSEQ*NHEAD-1
    const int n = slice >> 3;
    const int h = slice & 7;

    __shared__ alignas(16) char smem[54784];
    _Float16* qs   = (_Float16*)(smem);            // [128][72]
    _Float16* ks   = (_Float16*)(smem + 18432);    // [128][72]
    _Float16* vst  = (_Float16*)(smem + 36864);    // [64][136]  (v transposed)
    float*    mskf = (float*)   (smem + 54272);    // [128]
    _Float16* ps   = (_Float16*)(smem);            // [128][136] overlays qs+ks

    const int tid  = threadIdx.x;
    const int lane = tid & 31;
    const int w    = tid >> 5;             // wave id: owns rows [w*16, w*16+16)
    const int lr   = lane & 15;
    const int hi   = lane >> 4;
    const long base = ((long)n * MSG) * EDIM + h * DK;

    // Stage q, k, v^T
#pragma unroll
    for (int c = 0; c < 4; ++c) {
        int idx = tid + c * 256;           // 0..1023
        int r   = idx >> 3;                // 0..127
        int sg  = (idx & 7) * 8;           // 0..56
        v8h qv = *(const v8h*)(Q  + base + (long)r * EDIM + sg);
        *(v8h*)(qs + r * 72 + sg) = qv;
        v8h kv = *(const v8h*)(Km + base + (long)r * EDIM + sg);
        *(v8h*)(ks + r * 72 + sg) = kv;
        v8h vv = *(const v8h*)(V  + base + (long)r * EDIM + sg);
#pragma unroll
        for (int i = 0; i < 8; ++i) vst[(sg + i) * 136 + r] = vv[i];
    }
    if (tid < 128) mskf[tid] = mask[n * MSG + tid] ? -1e9f : 0.0f;
    __syncthreads();

    // scores = q @ k^T   (K-dim = 64 -> 2 WMMA steps), wave w: 16 rows x 128
    v8f sacc[8];
#pragma unroll
    for (int j = 0; j < 8; ++j)
#pragma unroll
        for (int t = 0; t < 8; ++t) sacc[j][t] = 0.0f;

#pragma unroll
    for (int kb = 0; kb < DK; kb += 32) {
        v16h a = ld_frag(qs + (w * 16 + lr) * 72, kb, hi);
#pragma unroll
        for (int j = 0; j < 8; ++j) {
            v16h b = ld_frag(ks + (j * 16 + lr) * 72, kb, hi);
            sacc[j] = __builtin_amdgcn_wmma_f32_16x16x32_f16(
                false, a, false, b, (short)0, sacc[j], false, false);
        }
    }
    __syncthreads();   // qs/ks dead after this point

    // masked softmax per row (row = w*16 + hi*8 + v, cols striped over 16 lanes)
#pragma unroll
    for (int v = 0; v < 8; ++v) {
        float rm = -3.0e38f;
#pragma unroll
        for (int j = 0; j < 8; ++j) {
            float val = sacc[j][v] * 0.125f + mskf[j * 16 + lr];
            sacc[j][v] = val;
            rm = fmaxf(rm, val);
        }
        rm = fmaxf(rm, __shfl_xor(rm, 1));
        rm = fmaxf(rm, __shfl_xor(rm, 2));
        rm = fmaxf(rm, __shfl_xor(rm, 4));
        rm = fmaxf(rm, __shfl_xor(rm, 8));
        float s = 0.0f;
#pragma unroll
        for (int j = 0; j < 8; ++j) {
            float e = __expf(sacc[j][v] - rm);
            sacc[j][v] = e;
            s += e;
        }
        s += __shfl_xor(s, 1);
        s += __shfl_xor(s, 2);
        s += __shfl_xor(s, 4);
        s += __shfl_xor(s, 8);
        float inv = 1.0f / s;
        int row = w * 16 + hi * 8 + v;
#pragma unroll
        for (int j = 0; j < 8; ++j)
            ps[row * 136 + j * 16 + lr] = (_Float16)(sacc[j][v] * inv);
    }
    __syncthreads();

    // o = p @ v   (K-dim = 128 -> 4 WMMA steps), wave w: 16 rows x 64 cols
    v8f oacc[4];
#pragma unroll
    for (int j = 0; j < 4; ++j)
#pragma unroll
        for (int t = 0; t < 8; ++t) oacc[j][t] = 0.0f;

#pragma unroll
    for (int kb = 0; kb < MSG; kb += 32) {
        v16h a = ld_frag(ps + (w * 16 + lr) * 136, kb, hi);
#pragma unroll
        for (int j = 0; j < 4; ++j) {
            v16h b = ld_frag(vst + (j * 16 + lr) * 136, kb, hi);
            oacc[j] = __builtin_amdgcn_wmma_f32_16x16x32_f16(
                false, a, false, b, (short)0, oacc[j], false, false);
        }
    }
#pragma unroll
    for (int j = 0; j < 4; ++j) {
        int gc = j * 16 + lr;
#pragma unroll
        for (int v = 0; v < 8; ++v) {
            int row = w * 16 + hi * 8 + v;
            O[base + (long)row * EDIM + gc] = (_Float16)oacc[j][v];
        }
    }
}

// ---------------------------------------------------------------------------
// AddNorm: out = x + LN(x), torch variant (unbiased std, eps added to std).
// One wave (32 lanes) per 512-wide row; 8 rows per 256-thread block.
// ---------------------------------------------------------------------------
__global__ void __launch_bounds__(256)
addnorm(const float* __restrict__ X, const float* __restrict__ w,
        const float* __restrict__ b, float* __restrict__ out) {
    const long row  = (long)blockIdx.x * 8 + (threadIdx.x >> 5);
    const int  lane = threadIdx.x & 31;
    const float* xr = X + row * EDIM;

    float4 x[4];
#pragma unroll
    for (int c = 0; c < 4; ++c)
        x[c] = *(const float4*)(xr + (lane + c * 32) * 4);

    float s = 0.0f;
#pragma unroll
    for (int c = 0; c < 4; ++c) s += x[c].x + x[c].y + x[c].z + x[c].w;
    s += __shfl_xor(s, 1);  s += __shfl_xor(s, 2);
    s += __shfl_xor(s, 4);  s += __shfl_xor(s, 8);  s += __shfl_xor(s, 16);
    const float mean = s * (1.0f / 512.0f);

    float ss = 0.0f;
#pragma unroll
    for (int c = 0; c < 4; ++c) {
        float dx = x[c].x - mean, dy = x[c].y - mean;
        float dz = x[c].z - mean, dw = x[c].w - mean;
        ss += dx * dx + dy * dy + dz * dz + dw * dw;
    }
    ss += __shfl_xor(ss, 1);  ss += __shfl_xor(ss, 2);
    ss += __shfl_xor(ss, 4);  ss += __shfl_xor(ss, 8);  ss += __shfl_xor(ss, 16);
    const float stdv = sqrtf(ss * (1.0f / 511.0f));   // ddof=1
    const float inv  = 1.0f / (stdv + 1e-6f);

#pragma unroll
    for (int c = 0; c < 4; ++c) {
        int col = (lane + c * 32) * 4;
        float4 wv = *(const float4*)(w + col);
        float4 bv = *(const float4*)(b + col);
        float4 o;
        o.x = x[c].x + (wv.x * (x[c].x - mean) * inv + bv.x);
        o.y = x[c].y + (wv.y * (x[c].y - mean) * inv + bv.y);
        o.z = x[c].z + (wv.z * (x[c].z - mean) * inv + bv.z);
        o.w = x[c].w + (wv.w * (x[c].w - mean) * inv + bv.w);
        *(float4*)(out + row * EDIM + col) = o;
    }
}

// ---------------------------------------------------------------------------
// Orchestration.  Workspace layout (bytes), with buffer reuse (536 MB total):
//   [0,   67M)  q  f16        -> later overlaid by attn f32 [0, 134M)
//   [67M, 134M) k  f16
//   [134M,201M) v  f16        -> later overlaid by x1 f32 [134M, 268M)
//   [201M,268M) o  f16 (attention out, pre-Wo)
//   [268M,536M) h  f16 (FFN hidden, 65536 x 2048)
//   [0,   134M) f  f32 (FFN out, overlays dead attn)
// ---------------------------------------------------------------------------
extern "C" void kernel_launch(void* const* d_in, const int* in_sizes, int n_in,
                              void* d_out, int out_size, void* d_ws, size_t ws_size,
                              hipStream_t stream) {
    const float* msg  = (const float*)d_in[0];
    const int*   mask = (const int*)  d_in[1];
    const float* Wq = (const float*)d_in[2];  const float* bq = (const float*)d_in[3];
    const float* Wk = (const float*)d_in[4];  const float* bk = (const float*)d_in[5];
    const float* Wv = (const float*)d_in[6];  const float* bv = (const float*)d_in[7];
    const float* Wo = (const float*)d_in[8];  const float* bo = (const float*)d_in[9];
    const float* ln1w = (const float*)d_in[10]; const float* ln1b = (const float*)d_in[11];
    const float* ln2w = (const float*)d_in[12]; const float* ln2b = (const float*)d_in[13];
    const float* w1 = (const float*)d_in[14]; const float* b1 = (const float*)d_in[15];
    const float* w2 = (const float*)d_in[16]; const float* b2 = (const float*)d_in[17];
    float* out = (float*)d_out;

    char* ws = (char*)d_ws;
    const size_t MB64 = (size_t)NROWS * EDIM * sizeof(_Float16);  // 67,108,864
    _Float16* q    = (_Float16*)(ws);
    _Float16* k    = (_Float16*)(ws + MB64);
    _Float16* v    = (_Float16*)(ws + 2 * MB64);
    _Float16* o    = (_Float16*)(ws + 3 * MB64);
    float*    attn = (float*)   (ws);                 // overlays q+k (dead)
    float*    x1   = (float*)   (ws + 2 * MB64);      // overlays v+o (dead)
    _Float16* hbuf = (_Float16*)(ws + 4 * MB64);      // 65536 x 2048 f16
    float*    fbuf = (float*)   (ws);                 // overlays attn (dead)

    dim3 blk(256);
    dim3 gE(NROWS / 128, EDIM / 256);   // (512, 2)
    dim3 gF(NROWS / 128, FDIM / 256);   // (512, 8)

    // Q/K/V projections (f32 in -> f16 out)
    gemm_wmma<float, false, _Float16><<<gE, blk, 0, stream>>>(msg, Wq, bq, q, EDIM, EDIM);
    gemm_wmma<float, false, _Float16><<<gE, blk, 0, stream>>>(msg, Wk, bk, k, EDIM, EDIM);
    gemm_wmma<float, false, _Float16><<<gE, blk, 0, stream>>>(msg, Wv, bv, v, EDIM, EDIM);

    // Fused masked attention per (seq, head)
    attn_wmma<<<NSEQ * NHEAD, blk, 0, stream>>>(q, k, v, mask, o);

    // Output projection (f16 in -> f32 out)
    gemm_wmma<_Float16, false, float><<<gE, blk, 0, stream>>>(o, Wo, bo, attn, EDIM, EDIM);

    // AddNorm 1
    addnorm<<<NROWS / 8, blk, 0, stream>>>(attn, ln1w, ln1b, x1);

    // FFN: relu(x1 @ w1 + b1) -> f16 hidden; then @ w2 + b2 -> f32
    gemm_wmma<float, true, _Float16><<<gF, blk, 0, stream>>>(x1, w1, b1, hbuf, EDIM, FDIM);
    gemm_wmma<_Float16, false, float><<<gE, blk, 0, stream>>>(hbuf, w2, b2, fbuf, FDIM, EDIM);

    // AddNorm 2 -> final output
    addnorm<<<NROWS / 8, blk, 0, stream>>>(fbuf, ln2w, ln2b, out);
}
